// TransformerEncoderLayer_23897198034988
// MI455X (gfx1250) — matrix operand order repaired and checked by reading, not compiled, not run on hardware
//
#include <hip/hip_runtime.h>
#include <hip/hip_bf16.h>
#include <cstdint>
#include <cstddef>

// ---------------------------------------------------------------- types
typedef __attribute__((ext_vector_type(16))) _Float16 v16h;
typedef __attribute__((ext_vector_type(8)))  _Float16 v8h;
typedef __attribute__((ext_vector_type(8)))  float    v8f;
typedef __attribute__((ext_vector_type(4)))  float    v4f;

#define B_   8
#define N_   1024
#define D_   1024
#define H_   16
#define HD_  64
#define DFF_ 4096
#define ROWS_ (B_ * N_)          // 8192
#define QKVW_ (3 * D_)           // 3072

// ---------------------------------------------------------------- helpers
__device__ __forceinline__ v8f zero8() {
  v8f z;
#pragma unroll
  for (int i = 0; i < 8; ++i) z[i] = 0.0f;
  return z;
}

__device__ __forceinline__ v16h cat8(v8h lo, v8h hi) {
  v16h r;
#pragma unroll
  for (int i = 0; i < 8; ++i) { r[i] = lo[i]; r[i + 8] = hi[i]; }
  return r;
}

__device__ __forceinline__ v8f wmma_f16(v16h a, v16h b, v8f c) {
  // D = A(16x32 f16) * B(32x16 f16) + C(16x16 f32)
  return __builtin_amdgcn_wmma_f32_16x16x32_f16(
      /*neg_a=*/false, a, /*neg_b=*/false, b,
      /*c_mod=*/(short)0, c, /*reuse_a=*/false, /*reuse_b=*/false);
}

// CDNA5 async copy: global -> LDS, 16 bytes per lane, tracked by ASYNCcnt.
__device__ __forceinline__ void async_copy_b128(unsigned lds_off,
                                                const void* gbase,
                                                unsigned byte_off) {
  uint64_t base = (uint64_t)(uintptr_t)gbase;
  asm volatile("global_load_async_to_lds_b128 %0, %1, %2 offset:0"
               :
               : "v"(lds_off), "v"(byte_off), "s"(base)
               : "memory");
}
__device__ __forceinline__ void ds_wait() {
  asm volatile("s_wait_dscnt 0x0" ::: "memory");
}

// ---------------------------------------------------------------- f32 -> f16
__global__ void k_f32_to_f16(const float* __restrict__ a,
                             _Float16* __restrict__ o, int n4) {
  int i = blockIdx.x * blockDim.x + threadIdx.x;
  int stride = gridDim.x * blockDim.x;
  for (; i < n4; i += stride) {
    v4f v = *(const v4f*)(a + (size_t)i * 4);
    v8h* dst = (v8h*)nullptr;  // silence unused-type warnings on some compilers
    (void)dst;
    _Float16 h0 = (_Float16)v[0], h1 = (_Float16)v[1];
    _Float16 h2 = (_Float16)v[2], h3 = (_Float16)v[3];
    _Float16* op = o + (size_t)i * 4;
    op[0] = h0; op[1] = h1; op[2] = h2; op[3] = h3;
  }
}

// ---------------------------------------------------------------- LayerNorm
// one block (256 threads) per row of D_=1024; outputs f16
__global__ void __launch_bounds__(256)
k_layernorm(const float* __restrict__ x, const float* __restrict__ g,
            const float* __restrict__ b, _Float16* __restrict__ y) {
  const int row  = blockIdx.x;
  const int tid  = threadIdx.x;
  const int lane = tid & 31;
  const int wave = tid >> 5;
  const float* xr = x + (size_t)row * D_;

  float v[4];
  float s = 0.0f, ss = 0.0f;
#pragma unroll
  for (int i = 0; i < 4; ++i) {
    v[i] = xr[tid + i * 256];
    s += v[i];
    ss += v[i] * v[i];
  }
#pragma unroll
  for (int m = 1; m < 32; m <<= 1) {
    s  += __shfl_xor(s,  m, 32);
    ss += __shfl_xor(ss, m, 32);
  }
  __shared__ float rs[8], rss[8];
  if (lane == 0) { rs[wave] = s; rss[wave] = ss; }
  __syncthreads();
  float ts = 0.0f, tss = 0.0f;
#pragma unroll
  for (int w = 0; w < 8; ++w) { ts += rs[w]; tss += rss[w]; }
  const float mu   = ts * (1.0f / D_);
  const float var  = tss * (1.0f / D_) - mu * mu;
  const float rstd = rsqrtf(var + 1e-5f);
#pragma unroll
  for (int i = 0; i < 4; ++i) {
    int c = tid + i * 256;
    y[(size_t)row * D_ + c] = (_Float16)((v[i] - mu) * rstd * g[c] + b[c]);
  }
}

// ---------------------------------------------------------------- GEMM
// C[M,N] = A[M,K](f16) * W[N,K](f16)^T + bias[N] (+ res[M,N])
// block: 256 thr = 8 waves (4 in M x 2 in N); block tile 128x128; K-step 64
// double-buffered LDS; async copies of tile t+1 overlap WMMA on tile t
template <typename OutT, bool ADD_RES>
__global__ void __launch_bounds__(256)
k_gemm(const _Float16* __restrict__ A, const _Float16* __restrict__ W,
       const float* __restrict__ bias, const float* __restrict__ res,
       OutT* __restrict__ C, int M, int N, int K) {
  __shared__ __align__(32) _Float16 sA[2][128 * 64];
  __shared__ __align__(32) _Float16 sB[2][128 * 64];

  const int tid  = threadIdx.x;
  const int lane = tid & 31;
  const int l15  = lane & 15;
  const int lhi  = lane >> 4;
  const int wave = tid >> 5;
  const int wm   = wave >> 1;  // 0..3
  const int wn   = wave & 1;   // 0..1
  const int rowBase = blockIdx.y * 128;
  const int colBase = blockIdx.x * 128;
  const int nT = K >> 6;       // number of 64-wide K tiles

  v8f acc[2][4];
#pragma unroll
  for (int mt = 0; mt < 2; ++mt)
#pragma unroll
    for (int nt = 0; nt < 4; ++nt) acc[mt][nt] = zero8();

  // stage one 128x64 tile pair into LDS buffer `buf` (8 async ops / wave)
  auto stage = [&](int tile, int buf) {
    const unsigned aoff = (unsigned)(uintptr_t)(&sA[buf][0]);
    const unsigned boff = (unsigned)(uintptr_t)(&sB[buf][0]);
    const int kb = tile * 64;
#pragma unroll
    for (int it = 0; it < 4; ++it) {
      int c  = it * 256 + tid;      // 0..1023
      int r  = c >> 3;              // 0..127
      int ko = (c & 7) * 8;         // 0..56
      unsigned goA = (unsigned)((((size_t)(rowBase + r)) * K + kb + ko) * 2);
      unsigned goB = (unsigned)((((size_t)(colBase + r)) * K + kb + ko) * 2);
      unsigned lo  = (unsigned)((r * 64 + ko) * 2);
      async_copy_b128(aoff + lo, A, goA);
      async_copy_b128(boff + lo, W, goB);
    }
  };

  stage(0, 0);
  for (int t = 0; t < nT; ++t) {
    const int buf = t & 1;
    if (t + 1 < nT) {
      stage(t + 1, buf ^ 1);  // prefetch next tile (overlaps compute)
      asm volatile("s_wait_asynccnt 0x8" ::: "memory");  // tile t done
    } else {
      asm volatile("s_wait_asynccnt 0x0" ::: "memory");
    }
    __syncthreads();  // tile t visible to all waves

    const _Float16* tA = &sA[buf][0];
    const _Float16* tB = &sB[buf][0];

    // A fragments: 2 row tiles x 2 K-chunks of 32
    v16h af[2][2];
#pragma unroll
    for (int mt = 0; mt < 2; ++mt) {
      int r = wm * 32 + mt * 16 + l15;
#pragma unroll
      for (int kc = 0; kc < 2; ++kc) {
        int off = kc * 32 + lhi * 16;
        v8h lo = *(const v8h*)&tA[r * 64 + off];
        v8h hi = *(const v8h*)&tA[r * 64 + off + 8];
        af[mt][kc] = cat8(lo, hi);
      }
    }
    // 4 col tiles x 2 K-chunks -> 16 WMMAs per K-step
#pragma unroll
    for (int nt = 0; nt < 4; ++nt) {
      int cn = wn * 64 + nt * 16 + l15;
#pragma unroll
      for (int kc = 0; kc < 2; ++kc) {
        v16h bf = *(const v16h*)&tB[cn * 64 + kc * 32 + lhi * 16];
#pragma unroll
        for (int mt = 0; mt < 2; ++mt)
          acc[mt][nt] = wmma_f16(af[mt][kc], bf, acc[mt][nt]);
      }
    }
    __syncthreads();  // readers done before next prefetch overwrites
  }

  // epilogue
#pragma unroll
  for (int mt = 0; mt < 2; ++mt)
#pragma unroll
    for (int nt = 0; nt < 4; ++nt)
#pragma unroll
      for (int r = 0; r < 8; ++r) {
        int row = rowBase + wm * 32 + mt * 16 + lhi * 8 + r;
        int col = colBase + wn * 64 + nt * 16 + l15;
        float v = acc[mt][nt][r] + bias[col];
        if (ADD_RES) v += res[(size_t)row * N + col];
        C[(size_t)row * N + col] = (OutT)v;
      }
}

// ---------------------------------------------------------------- attention
// flash-attention with ALiBi; qkv f16 [B*N, 3D]; out f16 [B*N, D]
// grid: (N/128, H, B); block 256 thr = 8 waves, 16 q-rows per wave
__global__ void __launch_bounds__(256)
k_attention(const _Float16* __restrict__ qkv, _Float16* __restrict__ out) {
  __shared__ __align__(32) _Float16 sK[64 * 64];
  __shared__ __align__(32) _Float16 sVt[64 * 64];     // transposed: [hd][key]
  __shared__ __align__(32) _Float16 sP[8][16 * 64];   // per-wave P staging

  const int tid  = threadIdx.x;
  const int lane = tid & 31;
  const int l15  = lane & 15;
  const int lhi  = lane >> 4;
  const int wave = tid >> 5;
  const int qt = blockIdx.x, h = blockIdx.y, b = blockIdx.z;

  const size_t rs = QKVW_;  // qkv row stride (halves)
  const _Float16* qbase = qkv + (size_t)b * N_ * rs + (size_t)h * HD_;
  const _Float16* kbase = qbase + D_;
  const _Float16* vbase = qbase + 2 * D_;
  const int q0 = qt * 128 + wave * 16;

  // load this wave's Q fragments once (2 K-chunks covering HD=64)
  v16h qa[2];
  {
    const _Float16* qp = qbase + (size_t)(q0 + l15) * rs;
    int off = lhi * 16;
#pragma unroll
    for (int c = 0; c < 2; ++c) {
      v8h lo = *(const v8h*)(qp + c * 32 + off);
      v8h hi = *(const v8h*)(qp + c * 32 + off + 8);
      qa[c] = cat8(lo, hi);
    }
  }

  const float slope = exp2f(-0.5f * (float)(h + 1));
  const float scale = 0.125f;  // 1/sqrt(64)

  v8f O[4];
#pragma unroll
  for (int t = 0; t < 4; ++t) O[t] = zero8();
  float Mrow[8], Lrow[8];
#pragma unroll
  for (int r = 0; r < 8; ++r) { Mrow[r] = -3.0e38f; Lrow[r] = 0.0f; }

  const unsigned sKoff = (unsigned)(uintptr_t)(&sK[0]);
  _Float16* pstage = &sP[wave][0];

  for (int kv = 0; kv < N_; kv += 64) {
    __syncthreads();
    // stage K tile (async) and V tile transposed (register scatter)
#pragma unroll
    for (int it = 0; it < 2; ++it) {
      int c   = it * 256 + tid;   // 0..511
      int key = c >> 3;           // 0..63
      int hd  = (c & 7) * 8;      // 0..56
      unsigned go = (unsigned)((((size_t)(kv + key)) * rs + hd) * 2);
      async_copy_b128(sKoff + (unsigned)((key * 64 + hd) * 2), kbase, go);
      v8h vv = *(const v8h*)(vbase + (size_t)(kv + key) * rs + hd);
#pragma unroll
      for (int i = 0; i < 8; ++i) sVt[(hd + i) * 64 + key] = vv[i];
    }
    asm volatile("s_wait_asynccnt 0x0" ::: "memory");
    __syncthreads();

    // S = Q K^T * scale + alibi, 4 sub-tiles of 16 keys
    v8f s[4];
#pragma unroll
    for (int st = 0; st < 4; ++st) {
      v8f a = zero8();
      int keyl = st * 16 + l15;
#pragma unroll
      for (int c = 0; c < 2; ++c) {
        v16h bf = *(const v16h*)&sK[keyl * 64 + c * 32 + lhi * 16];
        a = wmma_f16(qa[c], bf, a);
      }
      int kcol = kv + keyl;
#pragma unroll
      for (int r = 0; r < 8; ++r) {
        int qrow = q0 + lhi * 8 + r;
        float bias = -slope * fabsf((float)(qrow - kcol));
        a[r] = a[r] * scale + bias;
      }
      s[st] = a;
    }

    // online softmax: group max over 64 keys, rescale running state
    float fac[8];
#pragma unroll
    for (int r = 0; r < 8; ++r) {
      float m = fmaxf(fmaxf(s[0][r], s[1][r]), fmaxf(s[2][r], s[3][r]));
#pragma unroll
      for (int mk = 1; mk < 16; mk <<= 1) m = fmaxf(m, __shfl_xor(m, mk, 32));
      float nM = fmaxf(Mrow[r], m);
      fac[r] = __expf(Mrow[r] - nM);
      Mrow[r] = nM;
      Lrow[r] *= fac[r];
    }
#pragma unroll
    for (int t = 0; t < 4; ++t)
#pragma unroll
      for (int r = 0; r < 8; ++r) O[t][r] *= fac[r];

    // P = exp(S - M): accumulate row sums, stage f16 P in per-wave LDS
#pragma unroll
    for (int st = 0; st < 4; ++st) {
#pragma unroll
      for (int r = 0; r < 8; ++r) {
        float p = __expf(s[st][r] - Mrow[r]);
        float ps = p;
#pragma unroll
        for (int mk = 1; mk < 16; mk <<= 1) ps += __shfl_xor(ps, mk, 32);
        Lrow[r] += ps;
        pstage[(lhi * 8 + r) * 64 + st * 16 + l15] = (_Float16)p;
      }
    }
    ds_wait();  // wave-local LDS RAW

    // O += P V  (2 K-chunks of 32 keys x 4 hd tiles)
#pragma unroll
    for (int c = 0; c < 2; ++c) {
      int off = lhi * 16;
      v8h lo = *(const v8h*)&pstage[l15 * 64 + c * 32 + off];
      v8h hi = *(const v8h*)&pstage[l15 * 64 + c * 32 + off + 8];
      v16h pa = cat8(lo, hi);
#pragma unroll
      for (int t = 0; t < 4; ++t) {
        v16h bf = *(const v16h*)&sVt[(t * 16 + l15) * 64 + c * 32 + lhi * 16];
        O[t] = wmma_f16(pa, bf, O[t]);
      }
    }
  }

  // normalize and write out (head-interleaved layout [row, h*64+hd])
#pragma unroll
  for (int t = 0; t < 4; ++t)
#pragma unroll
    for (int r = 0; r < 8; ++r) {
      int qrow = q0 + lhi * 8 + r;
      float v = O[t][r] / Lrow[r];
      out[((size_t)b * N_ + qrow) * D_ + h * HD_ + t * 16 + l15] = (_Float16)v;
    }
}

// ---------------------------------------------------------------- SwiGLU
__global__ void k_swiglu(const _Float16* gate, const _Float16* up,
                         _Float16* o, int n) {
  int i = blockIdx.x * blockDim.x + threadIdx.x;
  int stride = gridDim.x * blockDim.x;
  for (; i < n; i += stride) {
    float g = (float)gate[i];
    float u = (float)up[i];
    o[i] = (_Float16)(g / (1.0f + __expf(-g)) * u);
  }
}

// ---------------------------------------------------------------- launcher
extern "C" void kernel_launch(void* const* d_in, const int* in_sizes, int n_in,
                              void* d_out, int out_size, void* d_ws,
                              size_t ws_size, hipStream_t stream) {
  const float* src    = (const float*)d_in[0];
  const float* wqkv   = (const float*)d_in[1];
  const float* bqkv   = (const float*)d_in[2];
  const float* wo     = (const float*)d_in[3];
  const float* bo     = (const float*)d_in[4];
  const float* g1     = (const float*)d_in[5];
  const float* b1     = (const float*)d_in[6];
  const float* g2     = (const float*)d_in[7];
  const float* b2     = (const float*)d_in[8];
  const float* w_gate = (const float*)d_in[9];
  const float* b_gate = (const float*)d_in[10];
  const float* w_up   = (const float*)d_in[11];
  const float* b_up   = (const float*)d_in[12];
  const float* w_down = (const float*)d_in[13];
  const float* b_down = (const float*)d_in[14];
  float* out = (float*)d_out;
  char* ws = (char*)d_ws;

  size_t o = 0;
  auto alloc = [&](size_t bytes) { size_t r = o; o += (bytes + 255) & ~(size_t)255; return r; };
  const size_t W16_QKV = alloc((size_t)QKVW_ * D_ * 2);
  const size_t W16_O   = alloc((size_t)D_ * D_ * 2);
  const size_t W16_G   = alloc((size_t)DFF_ * D_ * 2);
  const size_t W16_U   = alloc((size_t)DFF_ * D_ * 2);
  const size_t W16_D   = alloc((size_t)D_ * DFF_ * 2);
  const size_t H16     = alloc((size_t)ROWS_ * D_ * 2);     // ln1, reused for ln2
  const size_t QKV16   = alloc((size_t)ROWS_ * QKVW_ * 2);  // reused: gate16 spans QKV16+ATTN16
  const size_t ATTN16  = alloc((size_t)ROWS_ * D_ * 2);
  const size_t UP16    = alloc((size_t)ROWS_ * DFF_ * 2);
  const size_t X1      = alloc((size_t)ROWS_ * D_ * 4);
  const size_t GATE16  = QKV16;  // 8192*4096*2 == qkv16 + attn16 regions

  _Float16* w16qkv = (_Float16*)(ws + W16_QKV);
  _Float16* w16o   = (_Float16*)(ws + W16_O);
  _Float16* w16g   = (_Float16*)(ws + W16_G);
  _Float16* w16u   = (_Float16*)(ws + W16_U);
  _Float16* w16d   = (_Float16*)(ws + W16_D);
  _Float16* h16    = (_Float16*)(ws + H16);
  _Float16* qkv16  = (_Float16*)(ws + QKV16);
  _Float16* attn16 = (_Float16*)(ws + ATTN16);
  _Float16* gate16 = (_Float16*)(ws + GATE16);
  _Float16* up16   = (_Float16*)(ws + UP16);
  float*    x1     = (float*)(ws + X1);

  // 1) weights -> f16 (vectorized: 4 elems/thread)
  k_f32_to_f16<<<512, 256, 0, stream>>>(wqkv,   w16qkv, QKVW_ * D_ / 4);
  k_f32_to_f16<<<512, 256, 0, stream>>>(wo,     w16o,   D_ * D_ / 4);
  k_f32_to_f16<<<512, 256, 0, stream>>>(w_gate, w16g,   DFF_ * D_ / 4);
  k_f32_to_f16<<<512, 256, 0, stream>>>(w_up,   w16u,   DFF_ * D_ / 4);
  k_f32_to_f16<<<512, 256, 0, stream>>>(w_down, w16d,   D_ * DFF_ / 4);

  // 2) ln1
  k_layernorm<<<ROWS_, 256, 0, stream>>>(src, g1, b1, h16);

  // 3) qkv = ln1 @ wqkv^T + bqkv   [8192 x 3072]
  k_gemm<_Float16, false><<<dim3(QKVW_ / 128, ROWS_ / 128), 256, 0, stream>>>(
      h16, w16qkv, bqkv, nullptr, qkv16, ROWS_, QKVW_, D_);

  // 4) attention with ALiBi
  k_attention<<<dim3(N_ / 128, H_, B_), 256, 0, stream>>>(qkv16, attn16);

  // 5) x1 = src + attn @ wo^T + bo
  k_gemm<float, true><<<dim3(D_ / 128, ROWS_ / 128), 256, 0, stream>>>(
      attn16, w16o, bo, src, x1, ROWS_, D_, D_);

  // 6) ln2
  k_layernorm<<<ROWS_, 256, 0, stream>>>(x1, g2, b2, h16);

  // 7) gate / up projections
  k_gemm<_Float16, false><<<dim3(DFF_ / 128, ROWS_ / 128), 256, 0, stream>>>(
      h16, w16g, b_gate, nullptr, gate16, ROWS_, DFF_, D_);
  k_gemm<_Float16, false><<<dim3(DFF_ / 128, ROWS_ / 128), 256, 0, stream>>>(
      h16, w16u, b_up, nullptr, up16, ROWS_, DFF_, D_);

  // 8) silu(gate) * up, in place
  k_swiglu<<<2048, 256, 0, stream>>>(gate16, up16, gate16, ROWS_ * DFF_);

  // 9) out = x1 + (gate*up) @ w_down^T + b_down
  k_gemm<float, true><<<dim3(D_ / 128, ROWS_ / 128), 256, 0, stream>>>(
      gate16, w16d, b_down, x1, out, ROWS_, D_, DFF_);
}
